// _ScaledDotProductAttention_40785009443536
// MI455X (gfx1250) — matrix operand order repaired and checked by reading, not compiled, not run on hardware
//
#include <hip/hip_runtime.h>
#include <math.h>

// ---------------------------------------------------------------------------
// Fused gated attention for MI455X (gfx1250, wave32, WMMA).
//   scores  = QK^T*scale + prev   (f16 WMMA 16x16x32, f32 accum; f32 to HBM)
//   weights = softmax(scores)     (f16 panel in LDS, f32 math)
//   out     = g * (weights @ V)   (f16 WMMA; A-fragments loaded pre-packed
//                                  from the f16 LDS panel via ds_load_b128)
// One workgroup (4 waves) owns a 16-row q panel of one (b,h). The 16x2048
// panel is kept in LDS as f16 (~64.5 KB) so scores/weights hit HBM once and
// 4 workgroups fit per 320 KB WGP (16 waves resident).
// ---------------------------------------------------------------------------

typedef __attribute__((ext_vector_type(16))) _Float16 v16h;
typedef __attribute__((ext_vector_type(8)))  _Float16 v8h;
typedef __attribute__((ext_vector_type(8)))  float    v8f;

#define NB   2
#define NH   16
#define SEQ  2048
#define DIM  64
#define PANEL_ROWS 16
#define ROWPH (SEQ + 16)                        // row stride in halves; 32B pad
                                                // -> 8-bank skew between rows
#define LDS_BYTES (PANEL_ROWS * ROWPH * 2)      // 66048 B of 320 KB

static __device__ __forceinline__ v8f wmma_f16(v16h a, v16h b, v8f c) {
  // 8 args: (neg_a, A, neg_b, B, c_mod, C, reuse_a, reuse_b)
  return __builtin_amdgcn_wmma_f32_16x16x32_f16(false, a, false, b,
                                                (short)0, c, false, false);
}

// Pack a per-lane A-fragment (16-bit A, 16x32 MxK layout) from f32 memory:
// lane L: M = L%16; elements 0..7 hold K = 8*(L/16)+0..7, elements 8..15
// hold K = 16 + 8*(L/16)+0..7. Caller passes the two 8-float groups.
static __device__ __forceinline__ v16h pack_a_f32(const float* p0, const float* p1) {
  float t[16];
  *reinterpret_cast<float4*>(t + 0)  = reinterpret_cast<const float4*>(p0)[0];
  *reinterpret_cast<float4*>(t + 4)  = reinterpret_cast<const float4*>(p0)[1];
  *reinterpret_cast<float4*>(t + 8)  = reinterpret_cast<const float4*>(p1)[0];
  *reinterpret_cast<float4*>(t + 12) = reinterpret_cast<const float4*>(p1)[1];
  v16h r;
#pragma unroll
  for (int i = 0; i < 16; ++i) r[i] = (_Float16)t[i];
  return r;
}

// A-fragment straight from an f16 LDS panel: two contiguous 16-byte chunks
// (ds_load_b128 each), no conversions.
static __device__ __forceinline__ v16h pack_a_f16(const _Float16* p0,
                                                  const _Float16* p1) {
  v8h lo = *reinterpret_cast<const v8h*>(p0);
  v8h hi = *reinterpret_cast<const v8h*>(p1);
  v16h r;
#pragma unroll
  for (int i = 0; i < 8; ++i) { r[i] = lo[i]; r[8 + i] = hi[i]; }
  return r;
}

__global__ void __launch_bounds__(128)
fused_gated_attention(const float* __restrict__ q, const float* __restrict__ k,
                      const float* __restrict__ v, const float* __restrict__ g,
                      const float* __restrict__ prev,
                      const float* __restrict__ scale_p,
                      float* __restrict__ out, float* __restrict__ wts,
                      float* __restrict__ scr, float* __restrict__ wu) {
  extern __shared__ _Float16 lds[];

  const int tid  = threadIdx.x;
  const int lane = tid & 31;
  const int wv   = tid >> 5;          // wave 0..3
  const int m    = lane & 15;         // N / M index within tile
  const int hi   = lane >> 4;         // half-wave select

  const int qTile = blockIdx.x & 127; // 128 q-panels of 16 rows
  const int bh    = blockIdx.x >> 7;  // 0..31 (b*H + h)
  const int q0    = qTile * PANEL_ROWS;
  const float scale = *scale_p;

  const size_t bh_ss = (size_t)bh * SEQ * (size_t)SEQ;
  const size_t bh_sd = (size_t)bh * SEQ * DIM;

  // ===== Phase 1: scores panel = Q K^T * scale + prev (f32->HBM, f16->LDS) ==
  const float* qrow = q + bh_sd + (size_t)(q0 + m) * DIM;
  const v16h a_lo = pack_a_f32(qrow +      8 * hi, qrow + 16 + 8 * hi);
  const v16h a_hi = pack_a_f32(qrow + 32 + 8 * hi, qrow + 48 + 8 * hi);

  const float* kb = k + bh_sd;        // k layout [bh][D][S] == B row-major
  for (int st = 0; st < 32; ++st) {   // each wave: 32 s-tiles = 512 columns
    const int s = wv * 512 + st * 16;
    v16h b_lo, b_hi;                  // B-fragment: K = (e&7) + 8*hi + 16*(e>>3), N = m
#pragma unroll
    for (int e = 0; e < 16; ++e) {
      const int d = (e & 7) + 8 * hi + 16 * (e >> 3);
      b_lo[e] = (_Float16)kb[(size_t)d        * SEQ + s + m];
      b_hi[e] = (_Float16)kb[(size_t)(d + 32) * SEQ + s + m];
    }
    v8f acc = {};
    acc = wmma_f16(a_lo, b_lo, acc);
    acc = wmma_f16(a_hi, b_hi, acc);
#pragma unroll
    for (int r = 0; r < 8; ++r) {     // C layout: row = r + 8*hi, col = m
      const int lr = r + 8 * hi;
      const size_t gidx = bh_ss + (size_t)(q0 + lr) * SEQ + (s + m);
      const float val = acc[r] * scale + prev[gidx];
      lds[lr * ROWPH + s + m] = (_Float16)val;
      scr[gidx] = val;                // materialize attn_scores in f32
    }
  }
  __syncthreads();

  // ===== Phase 2: row softmax on the f16 panel (8 threads/row, f32 math) ====
  {
    const int row = tid >> 3;         // 0..15
    v8h* lrow = reinterpret_cast<v8h*>(lds + row * ROWPH + (tid & 7) * 256);
    float mx = -3.402823466e38f;
    for (int c = 0; c < 32; ++c) {    // 32 chunks x 8 halves = 256 elements
      v8h x = lrow[c];
#pragma unroll
      for (int i = 0; i < 8; ++i) mx = fmaxf(mx, (float)x[i]);
    }
#pragma unroll
    for (int o = 1; o < 8; o <<= 1) mx = fmaxf(mx, __shfl_xor(mx, o, 32));
    float sum = 0.f;
    for (int c = 0; c < 32; ++c) {
      v8h x = lrow[c];
      v8h e8;
#pragma unroll
      for (int i = 0; i < 8; ++i) {
        float e = __expf((float)x[i] - mx);
        sum += e;
        e8[i] = (_Float16)e;
      }
      lrow[c] = e8;
    }
#pragma unroll
    for (int o = 1; o < 8; o <<= 1) sum += __shfl_xor(sum, o, 32);
    const float inv = 1.f / sum;
    for (int c = 0; c < 32; ++c) {
      v8h x = lrow[c];
#pragma unroll
      for (int i = 0; i < 8; ++i) x[i] = (_Float16)((float)x[i] * inv);
      lrow[c] = x;
    }
  }
  __syncthreads();

  // Coalesced f32 global write of the normalized weights panel.
  {
    float* wbase = wts + bh_ss + (size_t)q0 * SEQ;
    for (int ci = tid; ci < PANEL_ROWS * (SEQ / 8); ci += 128) {
      const int r = ci >> 8;                 // 256 chunks per row
      const int c = (ci & 255) * 8;
      v8h x = *reinterpret_cast<const v8h*>(lds + r * ROWPH + c);
      float4 f0, f1;
      f0.x = (float)x[0]; f0.y = (float)x[1]; f0.z = (float)x[2]; f0.w = (float)x[3];
      f1.x = (float)x[4]; f1.y = (float)x[5]; f1.z = (float)x[6]; f1.w = (float)x[7];
      float4* dst = reinterpret_cast<float4*>(wbase + (size_t)r * SEQ + c);
      dst[0] = f0; dst[1] = f1;
    }
  }

  // ===== Phase 3: out = g * (W @ V); wave wv owns d-tile [wv*16, wv*16+16) ==
  {
    const int d0 = wv * 16;
    const float* vb = v + bh_sd;             // [S][D] row-major
    const _Float16* wrow = lds + m * ROWPH;  // A rows pre-packed in LDS
    v8f acc = {};
    for (int s = 0; s < SEQ; s += 32) {      // 64 f16 WMMA steps, K=32 each
      const v16h a = pack_a_f16(wrow + s + 8 * hi, wrow + s + 16 + 8 * hi);
      v16h b;
#pragma unroll
      for (int e = 0; e < 16; ++e)
        b[e] = (_Float16)vb[(size_t)(s + (e & 7) + 8 * hi + 16 * (e >> 3)) * DIM + d0 + m];
      acc = wmma_f16(a, b, acc);
      __builtin_prefetch(vb + (size_t)(s + 64) * DIM + d0 + m, 0, 0);
    }
#pragma unroll
    for (int r = 0; r < 8; ++r) {
      const size_t o = bh_sd + (size_t)(q0 + r + 8 * hi) * DIM + d0 + m;
      out[o] = acc[r] * g[o];
    }
  }

  if (blockIdx.x == 0 && tid == 0) wu[0] = 0.f;   // weight_u = zeros((1,))
}

extern "C" void kernel_launch(void* const* d_in, const int* in_sizes, int n_in,
                              void* d_out, int out_size, void* d_ws, size_t ws_size,
                              hipStream_t stream) {
  (void)in_sizes; (void)n_in; (void)out_size; (void)d_ws; (void)ws_size;
  const float* q     = (const float*)d_in[0];
  const float* k     = (const float*)d_in[1];
  const float* v     = (const float*)d_in[2];
  const float* g     = (const float*)d_in[3];
  const float* prev  = (const float*)d_in[4];
  const float* scale = (const float*)d_in[5];

  float* out = (float*)d_out;                       // [B,H,S,D]
  const size_t n_out  = (size_t)NB * NH * SEQ * DIM;
  const size_t n_attn = (size_t)NB * NH * SEQ * (size_t)SEQ;
  float* wts = out + n_out;                         // attn_weights [B,H,S,S]
  float* scr = wts + n_attn;                        // attn_scores  [B,H,S,S]
  float* wu  = scr + n_attn;                        // weight_u (1,)

  dim3 grid(NB * NH * (SEQ / PANEL_ROWS));          // 4096 workgroups
  dim3 block(128);                                  // 4 waves (wave32)
  hipLaunchKernelGGL(fused_gated_attention, grid, block, (size_t)LDS_BYTES,
                     stream, q, k, v, g, prev, scale, out, wts, scr, wu);
}
